// GNN_88072599372051
// MI455X (gfx1250) — matrix operand order repaired and checked by reading, not compiled, not run on hardware
//
#include <hip/hip_runtime.h>
#include <math.h>

// ---------------------------------------------------------------------------
// GraphNet (3-layer GN + readout) for MI455X / gfx1250, wave32 + WMMA.
//
// Memory-bound workload (edge-MLP gathers ~0.5GB/layer vs ~16 GFLOP/layer at
// 23.3 TB/s HBM) -> fuse gather + Linear + ReLU + Linear + bias + residual
// into one kernel. Gathers are staged coalesced into LDS (f16), then the
// matrix math runs on v_wmma_f32_16x16x32_f16 with pure ds_load_b128 feeds
// (no per-element control flow in the WMMA loop).
// ---------------------------------------------------------------------------

typedef __attribute__((ext_vector_type(16))) _Float16 v16h;
typedef __attribute__((ext_vector_type(8)))  _Float16 v8h;
typedef __attribute__((ext_vector_type(8)))  float    v8f;

#define GN_HID    64
#define GN_NN     50000
#define GN_NE     400000
#define GN_NG     256
#define GN_MAXSEG 6
#define GN_MAXCH  10          // max Kpad/32 (node layers: K=320 -> 10 chunks)
#define NEG_INF_F (-3.402823466e38f)

struct GnSeg {
  const float* data;   // feature matrix, row-major [rows][width]
  const int*   idx;    // nullptr => row index = m (direct)
  const int*   idx2;   // if non-null: row = idx2[idx[m]]  (e.g. batch[row[m]])
  int          width;
};

struct GnMlpArgs {
  GnSeg seg[GN_MAXSEG];
  int nseg, M, K1, Kpad;       // K1 = true input dim, Kpad = multiple of 32
  const float* W1;             // [K1][64] row-major f32
  const float* b1;             // [64]
  const float* W2;             // [64][64]
  const float* b2;             // [64]
  const float* res;            // residual (nullable; may alias out - safe)
  float*       out;            // [M][64]
  int relu2;                   // relu after second linear (readout)
};

// LDS offset (in halves) of a B-fragment element for V_WMMA_F32_16X16X32_F16:
// lane l holds column c=l%16, K=(l>=16?8:0)+(pos&7)+(pos>=8?16:0) per
// lane -> each 16x16 B tile is two ds_load_b128 per lane.
__device__ __forceinline__ int gn_frag_off(int chunk, int t, int lane, int pos) {
  return ((chunk * 4 + t) * 32 + lane) * 16 + pos;
}

__launch_bounds__(128)
__global__ void gn_fused_mlp2(GnMlpArgs A) {
  // LDS: A tile 40KB + W1 frags 40KB + W2 frags 8KB + hidden 8KB + rows 1.5KB
  __shared__ __align__(16) _Float16 sA [64 * (GN_MAXCH * 32)];
  __shared__ __align__(16) _Float16 sW1[GN_MAXCH * 2048];
  __shared__ __align__(16) _Float16 sW2[2 * 2048];
  __shared__ __align__(16) _Float16 sH [4 * 16 * GN_HID];
  __shared__ int sRow[GN_MAXSEG * 64];

  const int tid  = threadIdx.x;
  const int lane = tid & 31;
  const int wave = tid >> 5;
  const int nch  = A.Kpad >> 5;
  const int Kp   = A.Kpad;

  // Prefetch residual rows we will read in the epilogue (global_prefetch_b8).
  if (A.res) {
    int mm = blockIdx.x * 64 + (tid >> 1);
    if (mm < A.M)
      __builtin_prefetch(A.res + (long long)mm * GN_HID + (tid & 1) * 32, 0, 1);
  }

  // ---- Phase 1: zero A tile, stage weights, resolve gather row indices ----
  {
    v8h z = {};
    for (int e = tid; e < 8 * Kp; e += 128) *(v8h*)&sA[e * 8] = z;
  }
  for (int e = tid; e < Kp * GN_HID; e += 128) {   // W1 -> f16 frag layout
    int k = e >> 6, n = e & 63;
    float v = (k < A.K1) ? A.W1[k * GN_HID + n] : 0.f;
    int kk = k & 31;
    int l  = ((kk & 8) ? 16 : 0) + (n & 15);
    sW1[gn_frag_off(k >> 5, n >> 4, l, (kk & 7) | ((kk >> 1) & 8))] = (_Float16)v;
  }
  for (int e = tid; e < GN_HID * GN_HID; e += 128) {  // W2 -> f16 frag layout
    int k = e >> 6, n = e & 63;
    float v = A.W2[k * GN_HID + n];
    int kk = k & 31;
    int l  = ((kk & 8) ? 16 : 0) + (n & 15);
    sW2[gn_frag_off(k >> 5, n >> 4, l, (kk & 7) | ((kk >> 1) & 8))] = (_Float16)v;
  }
  for (int t = tid; t < A.nseg * 64; t += 128) {   // resolve row indices once
    int s = t >> 6, rl = t & 63;
    int m = blockIdx.x * 64 + rl;
    int r = (m < A.M) ? m : (A.M - 1);
    if (A.seg[s].idx)  r = A.seg[s].idx[r];
    if (A.seg[s].idx2) r = A.seg[s].idx2[r];
    sRow[s * 64 + rl] = r;
  }
  __syncthreads();

  // ---- Phase 2: coalesced gather of the virtual concat into sA (f16) ----
  {
    int ks = 0;
    for (int s = 0; s < A.nseg; ++s) {
      const float* dat = A.seg[s].data;
      const int    ws  = A.seg[s].width;
      for (int e = tid; e < 64 * ws; e += 128) {
        int rl = e / ws;
        int c  = e - rl * ws;
        float v = dat[(long long)sRow[s * 64 + rl] * ws + c];
        sA[rl * Kp + ks + c] = (_Float16)v;
      }
      ks += ws;
    }
  }
  __syncthreads();

  const int m0  = blockIdx.x * 64 + wave * 16;  // 16 rows per wave
  const int hib = (lane & 16) ? 8 : 0;          // A/C layout half-wave split
  const int arow = wave * 16 + (lane & 15);     // A-frag: M = lane%16

  union AF { v16h v; v8h u[2]; };
  union BF { v16h v; v8h u[2]; };

  // ---- GEMM1: Kp x 64, A frag = two ds_load_b128 from sA ----
  v8f acc[4] = {};
  for (int kc = 0; kc < nch; ++kc) {
    AF a;
    int ha = arow * Kp + kc * 32 + hib;   // two contiguous 8-half runs
    a.u[0] = *(const v8h*)&sA[ha];
    a.u[1] = *(const v8h*)&sA[ha + 16];
    BF b[4];
#pragma unroll
    for (int t = 0; t < 4; ++t) {
      int o = gn_frag_off(kc, t, lane, 0);
      b[t].u[0] = *(const v8h*)&sW1[o];
      b[t].u[1] = *(const v8h*)&sW1[o + 8];
    }
#pragma unroll
    for (int t = 0; t < 4; ++t)
      acc[t] = __builtin_amdgcn_wmma_f32_16x16x32_f16(
          false, a.v, false, b[t].v, (short)0, acc[t], false, false);
  }

  // ---- bias + ReLU, stage hidden (16x64 f16) per-wave ----
  // C layout: VGPR r -> row = r + hib, col = t*16 + lane%16.
  _Float16* hloc = &sH[wave * 16 * GN_HID];
#pragma unroll
  for (int t = 0; t < 4; ++t) {
    float bias = A.b1[t * 16 + (lane & 15)];
#pragma unroll
    for (int r = 0; r < 8; ++r) {
      float x = acc[t][r] + bias;
      x = x > 0.f ? x : 0.f;
      hloc[(r + hib) * GN_HID + t * 16 + (lane & 15)] = (_Float16)x;
    }
  }
  __syncthreads();

  // ---- GEMM2: hidden(16x64) @ W2(64x64) ----
  v8f acc2[4] = {};
#pragma unroll
  for (int kc = 0; kc < 2; ++kc) {
    AF a;
    int ha = (lane & 15) * GN_HID + kc * 32 + hib;
    a.u[0] = *(const v8h*)&hloc[ha];
    a.u[1] = *(const v8h*)&hloc[ha + 16];
    BF b[4];
#pragma unroll
    for (int t = 0; t < 4; ++t) {
      int o = gn_frag_off(kc, t, lane, 0);
      b[t].u[0] = *(const v8h*)&sW2[o];
      b[t].u[1] = *(const v8h*)&sW2[o + 8];
    }
#pragma unroll
    for (int t = 0; t < 4; ++t)
      acc2[t] = __builtin_amdgcn_wmma_f32_16x16x32_f16(
          false, a.v, false, b[t].v, (short)0, acc2[t], false, false);
  }

  // ---- Epilogue: bias, optional ReLU, optional residual (rows owned
  // exclusively by this wave -> in-place residual is race-free), store f32 ----
#pragma unroll
  for (int t = 0; t < 4; ++t) {
    int n = t * 16 + (lane & 15);
    float bias = A.b2[n];
#pragma unroll
    for (int r = 0; r < 8; ++r) {
      int m = m0 + r + hib;
      if (m < A.M) {
        float x = acc2[t][r] + bias;
        if (A.relu2) x = x > 0.f ? x : 0.f;
        if (A.res)   x += A.res[(long long)m * GN_HID + n];
        A.out[(long long)m * GN_HID + n] = x;
      }
    }
  }
}

// --------------------------- reduction kernels -----------------------------

__global__ void gn_fill_f32(float* p, int n, float v) {
  int i = blockIdx.x * blockDim.x + threadIdx.x;
  for (; i < n; i += gridDim.x * blockDim.x) p[i] = v;
}

__device__ __forceinline__ void gn_atomic_max_f32(float* addr, float val) {
  unsigned int* ua  = reinterpret_cast<unsigned int*>(addr);
  unsigned int  old = __float_as_uint(*addr);
  while (__uint_as_float(old) < val) {
    unsigned int assumed = old;
    old = atomicCAS(ua, assumed, __float_as_uint(val));
    if (old == assumed) break;
  }
}

// one thread per (element, feature): scatter add/max + count
__global__ void gn_scatter_stats(const float* __restrict__ e,
                                 const int* __restrict__ dst, int ne,
                                 float* s, float* mx, float* cnt) {
  int i = blockIdx.x * blockDim.x + threadIdx.x;
  if (i >= ne * GN_HID) return;
  int el = i >> 6, f = i & 63;
  int d  = dst[el];
  float v = e[i];
  atomicAdd(&s[(long long)d * GN_HID + f], v);
  gn_atomic_max_f32(&mx[(long long)d * GN_HID + f], v);
  if (f == 0) atomicAdd(&cnt[d], 1.0f);
}

__global__ void gn_finalize_stats(const float* __restrict__ s, float* mx,
                                  float* mean, const float* __restrict__ cnt,
                                  int n) {
  int i = blockIdx.x * blockDim.x + threadIdx.x;
  if (i >= n * GN_HID) return;
  float c = cnt[i >> 6];
  mean[i] = s[i] / fmaxf(c, 1.f);
  if (c <= 0.f) mx[i] = 0.f;   // torch_scatter convention: empty max -> 0
}

// readout l3: [256 x 64] @ [64 x 1] + b, sigmoid
__global__ void gn_l3_sigmoid(const float* __restrict__ o,
                              const float* __restrict__ W,
                              const float* __restrict__ b, float* out, int n) {
  int r = blockIdx.x * blockDim.x + threadIdx.x;
  if (r >= n) return;
  float acc = b[0];
  for (int k = 0; k < GN_HID; ++k) acc += o[r * GN_HID + k] * W[k];
  out[r] = 1.f / (1.f + expf(-acc));
}

// ------------------------------- launcher ----------------------------------

extern "C" void kernel_launch(void* const* d_in, const int* in_sizes, int n_in,
                              void* d_out, int out_size, void* d_ws,
                              size_t ws_size, hipStream_t stream) {
  // Locate tensors by unique flat sizes (robust to pytree flatten ordering).
  int i_batch = -1, i_eattr = -1, i_eidx = -1, i_u = -1, i_x = -1, i_p0 = -1;
  for (int i = 0; i < n_in; ++i) {
    int s = in_sizes[i];
    if (s == GN_NN) i_batch = i;
    else if (s == GN_NE * 4) i_eattr = i;
    else if (s == 2 * GN_NE) i_eidx = i;
    else if (s == GN_NG * 3) i_u = i;
    else if (s == GN_NN * 9) i_x = i;
    else if (s == 25 * 64 && i_p0 < 0) i_p0 = i;  // layers[0].edge.l1.W
  }
  if (i_batch < 0 || i_eattr < 0 || i_eidx < 0 || i_u < 0 || i_x < 0 || i_p0 < 0)
    return;

  const int*   batch = (const int*)d_in[i_batch];
  const int*   row   = (const int*)d_in[i_eidx];
  const int*   col   = row + GN_NE;
  const float* eattr = (const float*)d_in[i_eattr];
  const float* u     = (const float*)d_in[i_u];
  const float* x     = (const float*)d_in[i_x];

  // Params contiguous in JAX sorted-key pytree order starting at i_p0:
  // per layer L (stride 12): edge{l1.W,l1.b,l2.W,l2.b} glob{..} node{..};
  // then out: l1.W,l1.b,l2.W,l2.b,l3.W,l3.b at +36..41.
  auto P = [&](int off) { return (const float*)d_in[i_p0 + off]; };

  // Workspace carve (~154 MB total)
  char* w = (char*)d_ws;
  auto carve = [&](size_t nf) {
    float* p = (float*)w;
    w += ((nf * sizeof(float) + 255) / 256) * 256;
    return p;
  };
  float* e64     = carve((size_t)GN_NE * GN_HID);
  float* h64     = carve((size_t)GN_NN * GN_HID);
  float* g64     = carve((size_t)GN_NG * GN_HID);
  float* ns_s    = carve((size_t)GN_NN * GN_HID);
  float* ns_mx   = carve((size_t)GN_NN * GN_HID);
  float* ns_mean = carve((size_t)GN_NN * GN_HID);
  float* ncnt    = carve(GN_NN);
  float* gs_s    = carve((size_t)GN_NG * GN_HID);
  float* gs_mx   = carve((size_t)GN_NG * GN_HID);
  float* gs_mean = carve((size_t)GN_NG * GN_HID);
  float* gcnt    = carve(GN_NG);
  float* obuf    = carve((size_t)GN_NG * GN_HID);

  auto FILL = [&](float* p, int n, float v) {
    gn_fill_f32<<<(n + 255) / 256, 256, 0, stream>>>(p, n, v);
  };
  auto MLP = [&](const GnMlpArgs& a) {
    gn_fused_mlp2<<<(a.M + 63) / 64, 128, 0, stream>>>(a);
  };

  for (int L = 0; L < 3; ++L) {
    int pb = L * 12;

    // ---- EdgeModel: e = MLP([h[row], h[col], e, g[batch[row]]]) (+res) ----
    {
      GnMlpArgs a{};
      if (L == 0) {
        a.seg[0] = {x, row, nullptr, 9};
        a.seg[1] = {x, col, nullptr, 9};
        a.seg[2] = {eattr, nullptr, nullptr, 4};
        a.seg[3] = {u, row, batch, 3};        // g[batch[row[m]]]
        a.nseg = 4; a.K1 = 25; a.Kpad = 32; a.res = nullptr;
      } else {
        a.seg[0] = {h64, row, nullptr, GN_HID};
        a.seg[1] = {h64, col, nullptr, GN_HID};
        a.seg[2] = {e64, nullptr, nullptr, GN_HID};
        a.seg[3] = {g64, row, batch, GN_HID};
        a.nseg = 4; a.K1 = 4 * GN_HID; a.Kpad = 4 * GN_HID; a.res = e64;
      }
      a.M = GN_NE;
      a.W1 = P(pb + 0); a.b1 = P(pb + 1); a.W2 = P(pb + 2); a.b2 = P(pb + 3);
      a.out = e64; a.relu2 = 0;
      MLP(a);
    }

    // ---- edge -> node segment stats (sum / max / mean over col) ----
    FILL(ns_s, GN_NN * GN_HID, 0.f);
    FILL(ns_mx, GN_NN * GN_HID, NEG_INF_F);
    FILL(ncnt, GN_NN, 0.f);
    gn_scatter_stats<<<(GN_NE * GN_HID + 255) / 256, 256, 0, stream>>>(
        e64, col, GN_NE, ns_s, ns_mx, ncnt);
    gn_finalize_stats<<<(GN_NN * GN_HID + 255) / 256, 256, 0, stream>>>(
        ns_s, ns_mx, ns_mean, ncnt, GN_NN);

    // ---- NodeModel: h = MLP([h, s, mx, mean, g[batch]]) (+res) ----
    {
      GnMlpArgs a{};
      if (L == 0) {
        a.seg[0] = {x, nullptr, nullptr, 9};
        a.seg[4] = {u, batch, nullptr, 3};
        a.K1 = 9 + 3 * GN_HID + 3; a.Kpad = 224; a.res = nullptr;
      } else {
        a.seg[0] = {h64, nullptr, nullptr, GN_HID};
        a.seg[4] = {g64, batch, nullptr, GN_HID};
        a.K1 = 5 * GN_HID; a.Kpad = 5 * GN_HID; a.res = h64;
      }
      a.seg[1] = {ns_s, nullptr, nullptr, GN_HID};
      a.seg[2] = {ns_mx, nullptr, nullptr, GN_HID};
      a.seg[3] = {ns_mean, nullptr, nullptr, GN_HID};
      a.nseg = 5; a.M = GN_NN;
      a.W1 = P(pb + 8); a.b1 = P(pb + 9); a.W2 = P(pb + 10); a.b2 = P(pb + 11);
      a.out = h64; a.relu2 = 0;
      MLP(a);
    }

    // ---- node -> graph segment stats ----
    FILL(gs_s, GN_NG * GN_HID, 0.f);
    FILL(gs_mx, GN_NG * GN_HID, NEG_INF_F);
    FILL(gcnt, GN_NG, 0.f);
    gn_scatter_stats<<<(GN_NN * GN_HID + 255) / 256, 256, 0, stream>>>(
        h64, batch, GN_NN, gs_s, gs_mx, gcnt);
    gn_finalize_stats<<<(GN_NG * GN_HID + 255) / 256, 256, 0, stream>>>(
        gs_s, gs_mx, gs_mean, gcnt, GN_NG);

    // ---- GlobalModel: g = MLP([g, nmean]) (+res) ----
    {
      GnMlpArgs a{};
      if (L == 0) {
        a.seg[0] = {u, nullptr, nullptr, 3};
        a.K1 = 3 + GN_HID; a.Kpad = 96; a.res = nullptr;
      } else {
        a.seg[0] = {g64, nullptr, nullptr, GN_HID};
        a.K1 = 2 * GN_HID; a.Kpad = 2 * GN_HID; a.res = g64;
      }
      a.seg[1] = {gs_mean, nullptr, nullptr, GN_HID};
      a.nseg = 2; a.M = GN_NG;
      a.W1 = P(pb + 4); a.b1 = P(pb + 5); a.W2 = P(pb + 6); a.b2 = P(pb + 7);
      a.out = g64; a.relu2 = 0;
      MLP(a);
    }
  }

  // ---- readout: o = relu(L2(relu(L1([s, mean, mx, g])))) ----
  // (gs_* already hold stats of the final h)
  {
    GnMlpArgs a{};
    a.seg[0] = {gs_s, nullptr, nullptr, GN_HID};
    a.seg[1] = {gs_mean, nullptr, nullptr, GN_HID};
    a.seg[2] = {gs_mx, nullptr, nullptr, GN_HID};
    a.seg[3] = {g64, nullptr, nullptr, GN_HID};
    a.nseg = 4; a.K1 = 4 * GN_HID; a.Kpad = 4 * GN_HID; a.M = GN_NG;
    a.W1 = P(36); a.b1 = P(37); a.W2 = P(38); a.b2 = P(39);
    a.res = nullptr; a.out = obuf; a.relu2 = 1;
    MLP(a);
  }
  gn_l3_sigmoid<<<1, 256, 0, stream>>>(obuf, P(40), P(41), (float*)d_out,
                                       GN_NG);
}
